// MultiScaleDWConvBlock_15633680957984
// MI455X (gfx1250) — matrix operand order repaired and checked by reading, not compiled, not run on hardware
//
#include <hip/hip_runtime.h>
#include <hip/hip_bf16.h>
#include <math.h>

// ---------------------------------------------------------------------------
// MultiScaleDWConvBlock for MI455X (gfx1250, wave32)
//   B=32, C=256, H=W=64.
//   K1: fused dw3/dw5/dw7 + BN + erf-GELU + avg -> mid (f16, halves traffic)
//   Kw: pw weights f32 -> f16 (128 KB, L2-resident)
//   K2: 1x1 conv as GEMM on v_wmma_f32_16x16x32_f16 (f32 accum) + pw-BN
//       + per-channel sumsq reduce for GRN
//   K3: GRN stats (Gx, channel mean, Nx)
//   K4: GRN affine + residual + erf-GELU (in-place on d_out)
// Workspace: midh (64MB f16) + pwh (128KB f16) + sumsq/nx (64KB f32)
// ---------------------------------------------------------------------------

typedef __attribute__((ext_vector_type(16))) _Float16 v16h;
typedef __attribute__((ext_vector_type(8)))  float    v8f;

#define Bn   32
#define Cn   256
#define Hn   64
#define Wn   64
#define HWn  4096               // 2^12
#define NTOT (Bn * Cn * HWn)    // 33,554,432
#define SBS  264                // padded LDS row stride (f16): 132 dwords -> bank 4n

__device__ __forceinline__ float gelu_erf(float v) {
    // torch nn.GELU default: 0.5*x*(1+erf(x/sqrt(2)))
    return 0.5f * v * (1.0f + erff(v * 0.70710678118654752f));
}

// ---------------------------------------------------------------------------
// Kernel 1: fused depthwise 3x3/5x5/7x7 + BN + GELU, averaged -> mid (f16)
// One block = one (b,c) plane 32x32 tile; halo 3 (for 7x7) => 38x38 LDS tile.
// ---------------------------------------------------------------------------
__global__ __launch_bounds__(256) void k_dwfused(
    const float* __restrict__ x,
    const float* __restrict__ w3,
    const float* __restrict__ bn3w, const float* __restrict__ bn3b,
    const float* __restrict__ bn3m, const float* __restrict__ bn3v,
    const float* __restrict__ w5,
    const float* __restrict__ bn5w, const float* __restrict__ bn5b,
    const float* __restrict__ bn5m, const float* __restrict__ bn5v,
    const float* __restrict__ w7,
    const float* __restrict__ bn7w, const float* __restrict__ bn7b,
    const float* __restrict__ bn7m, const float* __restrict__ bn7v,
    _Float16* __restrict__ mid)
{
    __shared__ float sT[38 * 38];   // haloed input tile
    __shared__ float sW[84];        // [0..8]=w3  [9..33]=w5  [34..82]=w7
    __shared__ float sBn[6];        // {s3,h3,s5,h5,s7,h7}

    const int tid   = threadIdx.x;
    const int bc    = blockIdx.z;          // b*C + c
    const int c     = bc & (Cn - 1);
    const int tileX = blockIdx.x;          // 0..1
    const int tileY = blockIdx.y;          // 0..1

    if (tid < 9)                 sW[tid]      = w3[c * 9  + tid];
    else if (tid < 34)           sW[tid]      = w5[c * 25 + (tid - 9)];
    else if (tid < 83)           sW[tid]      = w7[c * 49 + (tid - 34)];
    else if (tid == 84) { float s = bn3w[c] * rsqrtf(bn3v[c] + 1e-5f);
                          sBn[0] = s; sBn[1] = bn3b[c] - bn3m[c] * s; }
    else if (tid == 85) { float s = bn5w[c] * rsqrtf(bn5v[c] + 1e-5f);
                          sBn[2] = s; sBn[3] = bn5b[c] - bn5m[c] * s; }
    else if (tid == 86) { float s = bn7w[c] * rsqrtf(bn7v[c] + 1e-5f);
                          sBn[4] = s; sBn[5] = bn7b[c] - bn7m[c] * s; }

    const int gy0 = tileY * 32 - 3;
    const int gx0 = tileX * 32 - 3;
    const float* __restrict__ xp = x + (size_t)bc * HWn;
    for (int i = tid; i < 38 * 38; i += 256) {
        const int ly = i / 38, lx = i - ly * 38;
        const int gy = gy0 + ly, gx = gx0 + lx;
        float v = 0.0f;
        if (gy >= 0 && gy < Hn && gx >= 0 && gx < Wn) v = xp[gy * Wn + gx];
        sT[i] = v;
    }
    __syncthreads();

    _Float16* __restrict__ mp = mid + (size_t)bc * HWn;
    for (int p = tid; p < 1024; p += 256) {
        const int oy = p >> 5, ox = p & 31;
        const float* ctr = &sT[(oy + 3) * 38 + (ox + 3)];

        float a3 = 0.0f;
        #pragma unroll
        for (int dy = -1; dy <= 1; ++dy)
            #pragma unroll
            for (int dx = -1; dx <= 1; ++dx)
                a3 = fmaf(ctr[dy * 38 + dx], sW[(dy + 1) * 3 + (dx + 1)], a3);

        float a5 = 0.0f;
        #pragma unroll
        for (int dy = -2; dy <= 2; ++dy)
            #pragma unroll
            for (int dx = -2; dx <= 2; ++dx)
                a5 = fmaf(ctr[dy * 38 + dx], sW[9 + (dy + 2) * 5 + (dx + 2)], a5);

        float a7 = 0.0f;
        #pragma unroll
        for (int dy = -3; dy <= 3; ++dy)
            #pragma unroll
            for (int dx = -3; dx <= 3; ++dx)
                a7 = fmaf(ctr[dy * 38 + dx], sW[34 + (dy + 3) * 7 + (dx + 3)], a7);

        const float v = gelu_erf(fmaf(a3, sBn[0], sBn[1]))
                      + gelu_erf(fmaf(a5, sBn[2], sBn[3]))
                      + gelu_erf(fmaf(a7, sBn[4], sBn[5]));
        mp[(tileY * 32 + oy) * Wn + (tileX * 32 + ox)] = (_Float16)(v * (1.0f / 3.0f));
    }
}

// ---------------------------------------------------------------------------
// weight convert f32 -> f16 (pw is 256x256, stays hot in L2)
// ---------------------------------------------------------------------------
__global__ void k_cvt_w(const float* __restrict__ pw, _Float16* __restrict__ pwh, int n) {
    const int i = blockIdx.x * blockDim.x + threadIdx.x;
    if (i < n) pwh[i] = (_Float16)pw[i];
}

// zero-init for the atomic sum-of-squares buffer (graph-safe, deterministic)
__global__ void k_zero(float* __restrict__ p, int n) {
    const int i = blockIdx.x * blockDim.x + threadIdx.x;
    if (i < n) p[i] = 0.0f;
}

// ---------------------------------------------------------------------------
// Kernel 2: pointwise 1x1 conv as f16 WMMA GEMM (f32 accum) + pw-BN + sumsq.
// Per batch b:  D[o, hw] = sum_c pw[o,c] * mid[b,c,hw]   (M=256, N=4096, K=256)
// Block = 512 threads = 16 waves; block owns (b, 16-column N panel);
// wave w owns M-tile rows [16w,16w+16). Panel staged TRANSPOSED in LDS:
// sB[n][c], row stride 264 f16 (132 dwords; 132 mod 64 = 4 -> rows hit
// distinct bank groups), so each lane's B fragment (two 8-K runs, same K as A) is two
// 16B ds loads.
//
// v_wmma_f32_16x16x32_f16 operand layout (ISA 7.12.2, 16-bit A 16x32):
//   A: lane<16 holds row M=lane, VGPR0-3 = K0..7,  VGPR4-7 = K16..23
//      lane>=16 holds row M=lane-16, VGPR0-3 = K8..15, VGPR4-7 = K24..31
//   B: lane holds column N=lane%16; lanes0-15 K0..7,K16..23; lanes16-31 K8..15,K24..31
//   C/D f32: VGPR r: lanes0-15 -> (M=r, N=lane), lanes16-31 -> (M=r+8)
// ---------------------------------------------------------------------------
__global__ __launch_bounds__(512) void k_pw_wmma(
    const _Float16* __restrict__ mid, const _Float16* __restrict__ pwh,
    const float* __restrict__ bnw, const float* __restrict__ bnb,
    const float* __restrict__ bnm, const float* __restrict__ bnv,
    float* __restrict__ out, float* __restrict__ sumsq)
{
    __shared__ _Float16 sB[16 * SBS];   // transposed activation panel (~8.3 KB)
    __shared__ float    sScale[Cn];
    __shared__ float    sShift[Cn];

    const int tid   = threadIdx.x;
    const int b     = blockIdx.x >> 8;          // 256 N-tiles per batch
    const int nBase = (blockIdx.x & 255) * 16;

    if (tid < Cn) {
        const float s = bnw[tid] * rsqrtf(bnv[tid] + 1e-5f);
        sScale[tid] = s;
        sShift[tid] = bnb[tid] - bnm[tid] * s;
    }

    // stage panel: read mid[c][nBase..nBase+15] as dwords, scatter transposed
    {
        const uint32_t* __restrict__ m32 =
            (const uint32_t*)(mid + (size_t)b * Cn * HWn);
        for (int i = tid; i < Cn * 8; i += 512) {      // 2048 dwords
            const int cc = i >> 3;
            const int np = (i & 7) * 2;                 // n pair
            union { uint32_t u; _Float16 h[2]; } w;
            w.u = m32[cc * (HWn / 2) + (nBase >> 1) + (i & 7)];
            sB[(np    ) * SBS + cc] = w.h[0];
            sB[(np + 1) * SBS + cc] = w.h[1];
        }
    }
    __syncthreads();

    const int wave  = tid >> 5;
    const int lane  = tid & 31;
    const int mBase = wave * 16;
    const int col   = lane & 15;
    const int hi    = (lane >= 16);
    const int aoff  = hi ? 8  : 0;      // A: K sub-offset per half-wave
    const int koff  = hi ? 16 : 0;      // unused; B loads use aoff
    const _Float16* __restrict__ pwRow = pwh + (size_t)(mBase + col) * Cn;
    const _Float16* __restrict__ bRow  = &sB[col * SBS];

    union AB { v16h v; float4 q[2]; };
    v8f acc = {0.f, 0.f, 0.f, 0.f, 0.f, 0.f, 0.f, 0.f};

    #pragma unroll
    for (int k = 0; k < Cn; k += 32) {
        AB a, bf;
        a.q[0]  = *(const float4*)(pwRow + k + aoff);        // K 0..7 / 8..15
        a.q[1]  = *(const float4*)(pwRow + k + aoff + 16);   // K 16..23 / 24..31
        bf.q[0] = *(const float4*)(bRow + k + aoff);         // K 0..7 / 8..15
        bf.q[1] = *(const float4*)(bRow + k + aoff + 16);    // K 16..23 / 24..31
        acc = __builtin_amdgcn_wmma_f32_16x16x32_f16(
                  /*neg_a=*/false, a.v, /*neg_b=*/false, bf.v,
                  /*c_mod=*/(short)0, acc, /*reuse_a=*/false, /*reuse_b=*/false);
    }

    // pw-BN, store, and per-channel sum-of-squares for GRN
    const int rowOff = hi ? 8 : 0;
    float* __restrict__ outb = out + (size_t)b * Cn * HWn;
    float* __restrict__ ssb  = sumsq + b * Cn;
    #pragma unroll
    for (int r = 0; r < 8; ++r) {
        const int o = mBase + r + rowOff;
        const float v = fmaf(acc[r], sScale[o], sShift[o]);
        outb[(size_t)o * HWn + nBase + col] = v;
        // xor masks < 16 never cross the 16-lane halves -> full-width shuffles
        float s = v * v;
        s += __shfl_xor(s, 1);
        s += __shfl_xor(s, 2);
        s += __shfl_xor(s, 4);
        s += __shfl_xor(s, 8);
        if (col == 0) unsafeAtomicAdd(&ssb[o], s);   // global_atomic_add_f32
    }
}

// ---------------------------------------------------------------------------
// Kernel 3: GRN stats. Gx[b,c]=sqrt(sumsq); Nx = Gx / (mean_c Gx + eps)
// ---------------------------------------------------------------------------
__global__ __launch_bounds__(256) void k_grn_stats(
    const float* __restrict__ sumsq, float* __restrict__ nx)
{
    __shared__ float red[256];
    const int b = blockIdx.x, c = threadIdx.x;
    const float g = sqrtf(sumsq[b * Cn + c]);
    red[c] = g;
    __syncthreads();
    #pragma unroll
    for (int s = 128; s > 0; s >>= 1) {
        if (c < s) red[c] += red[c + s];
        __syncthreads();
    }
    const float mean = red[0] * (1.0f / Cn);
    nx[b * Cn + c] = g / (mean + 1e-6f);
}

// ---------------------------------------------------------------------------
// Kernel 4: out = gelu(out + gamma*(out*Nx) + beta + residual), in place.
// float4 path; HW=4096 divisible by 4 so a float4 never crosses a channel.
// ---------------------------------------------------------------------------
__global__ __launch_bounds__(256) void k_grn_apply(
    const float* __restrict__ x, const float* __restrict__ nx,
    const float* __restrict__ gamma, const float* __restrict__ beta,
    float* __restrict__ out)
{
    const size_t base = ((size_t)blockIdx.x * blockDim.x + threadIdx.x) * 4;
    const int b = (int)(base >> 20);          // C*HW = 2^20 elems per batch
    const int c = (int)((base >> 12) & 255);  // HW = 2^12
    const float n  = nx[b * Cn + c];
    const float gm = gamma[c], bt = beta[c];

    const float4 t = *(const float4*)(out + base);
    const float4 r = *(const float4*)(x + base);
    float4 o;
    o.x = gelu_erf(t.x + gm * (t.x * n) + bt + r.x);
    o.y = gelu_erf(t.y + gm * (t.y * n) + bt + r.y);
    o.z = gelu_erf(t.z + gm * (t.z * n) + bt + r.z);
    o.w = gelu_erf(t.w + gm * (t.w * n) + bt + r.w);
    *(float4*)(out + base) = o;
}

// ---------------------------------------------------------------------------
extern "C" void kernel_launch(void* const* d_in, const int* in_sizes, int n_in,
                              void* d_out, int out_size, void* d_ws, size_t ws_size,
                              hipStream_t stream)
{
    (void)in_sizes; (void)n_in; (void)out_size; (void)ws_size;

    const float* x     = (const float*)d_in[0];
    const float* dw3   = (const float*)d_in[1];
    const float* bn3w  = (const float*)d_in[2];
    const float* bn3b  = (const float*)d_in[3];
    const float* bn3m  = (const float*)d_in[4];
    const float* bn3v  = (const float*)d_in[5];
    const float* dw5   = (const float*)d_in[6];
    const float* bn5w  = (const float*)d_in[7];
    const float* bn5b  = (const float*)d_in[8];
    const float* bn5m  = (const float*)d_in[9];
    const float* bn5v  = (const float*)d_in[10];
    const float* dw7   = (const float*)d_in[11];
    const float* bn7w  = (const float*)d_in[12];
    const float* bn7b  = (const float*)d_in[13];
    const float* bn7m  = (const float*)d_in[14];
    const float* bn7v  = (const float*)d_in[15];
    const float* pw    = (const float*)d_in[16];
    const float* pbw   = (const float*)d_in[17];
    const float* pbb   = (const float*)d_in[18];
    const float* pbm   = (const float*)d_in[19];
    const float* pbv   = (const float*)d_in[20];
    const float* ggam  = (const float*)d_in[21];
    const float* gbet  = (const float*)d_in[22];

    float* out      = (float*)d_out;
    _Float16* midh  = (_Float16*)d_ws;                 // 64 MB
    _Float16* pwh   = midh + (size_t)NTOT;             // 128 KB
    float*    sumsq = (float*)(pwh + Cn * Cn);         // 32 KB
    float*    nx    = sumsq + Bn * Cn;                 // 32 KB

    // 1) fused multi-scale depthwise + BN + GELU -> midh (f16)
    dim3 g1(Wn / 32, Hn / 32, Bn * Cn);
    k_dwfused<<<g1, 256, 0, stream>>>(x,
        dw3, bn3w, bn3b, bn3m, bn3v,
        dw5, bn5w, bn5b, bn5m, bn5v,
        dw7, bn7w, bn7b, bn7m, bn7v, midh);

    // weight convert + sumsq zero (independent of K1 output)
    k_cvt_w<<<(Cn * Cn + 255) / 256, 256, 0, stream>>>(pw, pwh, Cn * Cn);
    k_zero<<<(Bn * Cn + 255) / 256, 256, 0, stream>>>(sumsq, Bn * Cn);

    // 2) f16 WMMA pointwise GEMM + pw-BN + sumsq (out = GEMM result)
    k_pw_wmma<<<Bn * (HWn / 16), 512, 0, stream>>>(midh, pwh, pbw, pbb, pbm, pbv,
                                                   out, sumsq);

    // 3) GRN stats
    k_grn_stats<<<Bn, 256, 0, stream>>>(sumsq, nx);

    // 4) GRN affine + residual + GELU, in place on d_out
    k_grn_apply<<<NTOT / 4 / 256, 256, 0, stream>>>(x, nx, ggam, gbet, out);
}
